// LSTMModel_14319420965124
// MI455X (gfx1250) — compile-verified, matrix operand order checked by reading, not run on hardware
//
#include <hip/hip_runtime.h>
#include <hip/hip_bf16.h>

// ---------------- problem constants ----------------
#define B_ROWS 4096
#define T_STEPS 40
#define E_DIM 50
#define H_DIM 200
#define VOCAB 50000
#define D_IN 250           // E + H (LSTM GEMM K, padded to 256)
#define KP_LSTM 256
#define KC_LSTM 8          // 256 / 32
#define N4H 800            // 4*H
#define NT_LSTM 50         // 800 / 16 N-tiles
#define HP 224             // H padded to multiple of 32
#define KC_PROJ 7          // 224 / 32
#define NT_PROJ 3125       // 50000 / 16 N-tiles

typedef __attribute__((ext_vector_type(16))) __bf16 v16bf;
typedef __attribute__((ext_vector_type(8)))  __bf16 v8bf;
typedef __attribute__((ext_vector_type(8)))  float  v8f;

static __device__ __forceinline__ unsigned short f2bf(float f) {
  unsigned int u = __float_as_uint(f);
  u += 0x7FFFu + ((u >> 16) & 1u);   // round-to-nearest-even
  return (unsigned short)(u >> 16);
}
// A/B fragment per-lane layout (16-bit, 16x16x32): lane holds row/col lane&15,
// K split: elements 0..7 -> K = kc*32 + (lane>>4)*8 + e, elements 8..15 -> +16.
// Two contiguous 16B halves -> two b128 loads.
static __device__ __forceinline__ v16bf load_frag16(const __bf16* p) {
  v8bf lo = *reinterpret_cast<const v8bf*>(p);
  v8bf hi = *reinterpret_cast<const v8bf*>(p + 16);
  return __builtin_shufflevector(lo, hi, 0,1,2,3,4,5,6,7,8,9,10,11,12,13,14,15);
}
static __device__ __forceinline__ float sigm(float x) { return 1.0f / (1.0f + __expf(-x)); }
static __device__ __forceinline__ float tanh_fast(float x) { return 2.0f * sigm(2.0f * x) - 1.0f; }

// ---------------- pack W (250x800 f32 -> bf16 B-fragment tiles, K padded 256) --------
__global__ void pack_w_kernel(const float* __restrict__ W, unsigned short* __restrict__ Wp) {
  int tid = blockIdx.x * blockDim.x + threadIdx.x;
  if (tid >= NT_LSTM * KC_LSTM * 32) return;
  int lane = tid & 31;
  int chunk = (tid >> 5) % KC_LSTM;
  int n = (tid / (32 * KC_LSTM)) * 16 + (lane & 15);
  int half = lane >> 4;
#pragma unroll
  for (int e = 0; e < 16; ++e) {
    int k = chunk * 32 + (e >> 3) * 16 + half * 8 + (e & 7);
    float v = (k < D_IN) ? W[(size_t)k * N4H + n] : 0.0f;
    Wp[(size_t)tid * 16 + e] = f2bf(v);
  }
}

// ---------------- pack U (200x50000 f32 -> bf16 B-fragment tiles, K padded 224) ------
__global__ void pack_u_kernel(const float* __restrict__ U, unsigned short* __restrict__ Up) {
  int tid = blockIdx.x * blockDim.x + threadIdx.x;
  if (tid >= NT_PROJ * KC_PROJ * 32) return;
  int lane = tid & 31;
  int chunk = (tid >> 5) % KC_PROJ;
  int n = (tid / (32 * KC_PROJ)) * 16 + (lane & 15);
  int half = lane >> 4;
#pragma unroll
  for (int e = 0; e < 16; ++e) {
    int k = chunk * 32 + (e >> 3) * 16 + half * 8 + (e & 7);
    float v = (k < H_DIM) ? U[(size_t)k * VOCAB + n] : 0.0f;
    Up[(size_t)tid * 16 + e] = f2bf(v);
  }
}

// ---------------- fused embedding-gather + 40-step LSTM ------------------------------
// One WG = 16 batch rows, 10 waves; per step each wave does 5 N-tiles x 8 K WMMAs.
__global__ __launch_bounds__(320, 1)
void lstm_kernel(const int* __restrict__ ids, const float* __restrict__ emb,
                 const float* __restrict__ bl, const unsigned short* __restrict__ Wp,
                 unsigned short* __restrict__ hOut) {
  __shared__ unsigned short Abuf[16 * KP_LSTM];  // [x_t | h | zero-pad] bf16
  __shared__ float zbuf[16 * N4H];
  __shared__ float bbuf[N4H];

  const int tid = threadIdx.x;
  const int wave = tid >> 5;
  const int lane = tid & 31;
  const int mrow = lane & 15;
  const int half = lane >> 4;
  const int rowBase = blockIdx.x * 16;

  for (int i = tid; i < 16 * KP_LSTM; i += 320) Abuf[i] = 0;  // h0 = 0, K pad = 0
  for (int i = tid; i < N4H; i += 320) bbuf[i] = bl[i];

  float c_st[10];
#pragma unroll
  for (int i = 0; i < 10; ++i) c_st[i] = 0.0f;

  const __bf16* ab = reinterpret_cast<const __bf16*>(Abuf);
  const v16bf* WpV = reinterpret_cast<const v16bf*>(Wp);

#pragma clang loop unroll(disable)
  for (int t = 0; t < T_STEPS; ++t) {
    __syncthreads();
    // gather x_t -> Abuf[:, 0:50]
    for (int i = tid; i < 16 * E_DIM; i += 320) {
      int r = i / E_DIM, cc = i % E_DIM;
      int tok = ids[(size_t)(rowBase + r) * T_STEPS + t];
      Abuf[r * KP_LSTM + cc] = f2bf(emb[(size_t)tok * E_DIM + cc]);
    }
    __syncthreads();

    v16bf aF[KC_LSTM];
#pragma unroll
    for (int kc = 0; kc < KC_LSTM; ++kc)
      aF[kc] = load_frag16(ab + mrow * KP_LSTM + kc * 32 + half * 8);

    // keep only ONE tile's B fragments live -> no spilling
#pragma clang loop unroll(disable)
    for (int i = 0; i < 5; ++i) {
      int tile = wave + 10 * i;
      v16bf bF[KC_LSTM];
#pragma unroll
      for (int kc = 0; kc < KC_LSTM; ++kc)
        bF[kc] = WpV[(tile * KC_LSTM + kc) * 32 + lane];
      v8f acc = {};
#pragma unroll
      for (int kc = 0; kc < KC_LSTM; ++kc)
        acc = __builtin_amdgcn_wmma_f32_16x16x32_bf16(false, aF[kc], false, bF[kc],
                                                      (short)0, acc, false, false);
#pragma unroll
      for (int v = 0; v < 8; ++v)
        zbuf[(v + 8 * half) * N4H + tile * 16 + mrow] = acc[v];
    }
    __syncthreads();

    // gates: i,j,f,o ; forget_bias = 1
#pragma unroll
    for (int i = 0; i < 10; ++i) {
      int idx = tid + i * 320;
      int r = idx / H_DIM, k = idx % H_DIM;
      float iv = zbuf[r * N4H + k] + bbuf[k];
      float jv = zbuf[r * N4H + 200 + k] + bbuf[200 + k];
      float fv = zbuf[r * N4H + 400 + k] + bbuf[400 + k];
      float ov = zbuf[r * N4H + 600 + k] + bbuf[600 + k];
      float cn = sigm(fv + 1.0f) * c_st[i] + sigm(iv) * tanh_fast(jv);
      c_st[i] = cn;
      Abuf[r * KP_LSTM + E_DIM + k] = f2bf(sigm(ov) * tanh_fast(cn));
    }
  }
  __syncthreads();
  for (int i = tid; i < 16 * HP; i += 320) {
    int r = i / HP, k = i % HP;
    hOut[(size_t)(rowBase + r) * HP + k] =
        (k < H_DIM) ? Abuf[r * KP_LSTM + E_DIM + k] : (unsigned short)0;
  }
}

// ---------------- projection + online softmax (recompute pass 2) ---------------------
// One WG = 32 batch rows (2 M-tiles reuse each B fragment), 8 waves stride N-tiles.
__global__ __launch_bounds__(256, 1)
void proj_kernel(const unsigned short* __restrict__ hB, const unsigned short* __restrict__ Up,
                 const float* __restrict__ b2, float* __restrict__ out) {
  __shared__ unsigned short Hbuf[32 * HP];
  __shared__ float redM[8 * 32];
  __shared__ float redS[8 * 32];
  __shared__ float rowM[32];
  __shared__ float rowInvS[32];

  const int tid = threadIdx.x;
  const int wave = tid >> 5;
  const int lane = tid & 31;
  const int mrow = lane & 15;
  const int half = lane >> 4;
  const int rowBase = blockIdx.x * 32;

  for (int i = tid; i < 32 * HP; i += 256) Hbuf[i] = hB[(size_t)rowBase * HP + i];
  __syncthreads();

  const __bf16* hbf = reinterpret_cast<const __bf16*>(Hbuf);
  const v16bf* UpV = reinterpret_cast<const v16bf*>(Up);

  v16bf aF[2][KC_PROJ];  // held in registers for both passes
#pragma unroll
  for (int mt = 0; mt < 2; ++mt)
#pragma unroll
    for (int kc = 0; kc < KC_PROJ; ++kc)
      aF[mt][kc] = load_frag16(hbf + (mt * 16 + mrow) * HP + kc * 32 + half * 8);

  float mRun[2][8], sRun[2][8];
#pragma unroll
  for (int mt = 0; mt < 2; ++mt)
#pragma unroll
    for (int v = 0; v < 8; ++v) { mRun[mt][v] = -1e30f; sRun[mt][v] = 0.0f; }

  // pass 1: logits -> running (max, sum(exp)) per (row, lane-column)
#pragma clang loop unroll(disable)
  for (int tile = wave; tile < NT_PROJ; tile += 8) {
    float bcol = b2[tile * 16 + mrow];
    v16bf bF[KC_PROJ];
#pragma unroll
    for (int kc = 0; kc < KC_PROJ; ++kc)
      bF[kc] = UpV[((size_t)tile * KC_PROJ + kc) * 32 + lane];
#pragma unroll
    for (int mt = 0; mt < 2; ++mt) {
      v8f acc = {};
#pragma unroll
      for (int kc = 0; kc < KC_PROJ; ++kc)
        acc = __builtin_amdgcn_wmma_f32_16x16x32_bf16(false, aF[mt][kc], false, bF[kc],
                                                      (short)0, acc, false, false);
#pragma unroll
      for (int v = 0; v < 8; ++v) {
        float x = acc[v] + bcol;
        float mN = fmaxf(mRun[mt][v], x);
        sRun[mt][v] = sRun[mt][v] * __expf(mRun[mt][v] - mN) + __expf(x - mN);
        mRun[mt][v] = mN;
      }
    }
  }

  // reduce across 16 lanes of each half-wave, then across 8 waves via LDS
#pragma unroll
  for (int mt = 0; mt < 2; ++mt)
#pragma unroll
    for (int v = 0; v < 8; ++v) {
      float m = mRun[mt][v], s = sRun[mt][v];
#pragma unroll
      for (int off = 1; off < 16; off <<= 1) {
        float mo = __shfl_xor(m, off, 32);
        float so = __shfl_xor(s, off, 32);
        float mN = fmaxf(m, mo);
        s = s * __expf(m - mN) + so * __expf(mo - mN);
        m = mN;
      }
      if (mrow == 0) {
        int row = mt * 16 + 8 * half + v;
        redM[wave * 32 + row] = m;
        redS[wave * 32 + row] = s;
      }
    }
  __syncthreads();
  if (tid < 32) {
    float m = -1e30f, s = 0.0f;
    for (int w = 0; w < 8; ++w) {
      float mo = redM[w * 32 + tid], so = redS[w * 32 + tid];
      float mN = fmaxf(m, mo);
      s = s * __expf(m - mN) + so * __expf(mo - mN);
      m = mN;
    }
    rowM[tid] = m;
    rowInvS[tid] = 1.0f / s;
  }
  __syncthreads();

  float fm[2][8], fs[2][8];
#pragma unroll
  for (int mt = 0; mt < 2; ++mt)
#pragma unroll
    for (int v = 0; v < 8; ++v) {
      int row = mt * 16 + 8 * half + v;
      fm[mt][v] = rowM[row];
      fs[mt][v] = rowInvS[row];
    }

  // pass 2: recompute logits, write probabilities.
  // d_out is write-once streaming data -> non-temporal stores keep packed-U in L2.
#pragma clang loop unroll(disable)
  for (int tile = wave; tile < NT_PROJ; tile += 8) {
    float bcol = b2[tile * 16 + mrow];
    v16bf bF[KC_PROJ];
#pragma unroll
    for (int kc = 0; kc < KC_PROJ; ++kc)
      bF[kc] = UpV[((size_t)tile * KC_PROJ + kc) * 32 + lane];
#pragma unroll
    for (int mt = 0; mt < 2; ++mt) {
      v8f acc = {};
#pragma unroll
      for (int kc = 0; kc < KC_PROJ; ++kc)
        acc = __builtin_amdgcn_wmma_f32_16x16x32_bf16(false, aF[mt][kc], false, bF[kc],
                                                      (short)0, acc, false, false);
#pragma unroll
      for (int v = 0; v < 8; ++v) {
        int row = mt * 16 + 8 * half + v;
        float val = __expf(acc[v] + bcol - fm[mt][v]) * fs[mt][v];
        __builtin_nontemporal_store(val,
            out + (size_t)(rowBase + row) * VOCAB + tile * 16 + mrow);
      }
    }
  }
}

// ---------------- launcher ------------------------------------------------------------
extern "C" void kernel_launch(void* const* d_in, const int* in_sizes, int n_in,
                              void* d_out, int out_size, void* d_ws, size_t ws_size,
                              hipStream_t stream) {
  (void)in_sizes; (void)n_in; (void)out_size; (void)ws_size;
  const int* ids = (const int*)d_in[0];
  const float* emb = (const float*)d_in[1];
  const float* W = (const float*)d_in[2];
  const float* bl = (const float*)d_in[3];
  const float* U = (const float*)d_in[4];
  const float* b2 = (const float*)d_in[5];
  float* out = (float*)d_out;

  // workspace layout (bf16/ushort), ~24.7 MB total
  unsigned short* Wp = (unsigned short*)d_ws;                         // 50*8*32*16
  unsigned short* Up = Wp + (size_t)NT_LSTM * KC_LSTM * 32 * 16;      // 3125*7*32*16
  unsigned short* hB = Up + (size_t)NT_PROJ * KC_PROJ * 32 * 16;      // 4096*224

  hipLaunchKernelGGL(pack_w_kernel, dim3((NT_LSTM * KC_LSTM * 32 + 255) / 256),
                     dim3(256), 0, stream, W, Wp);
  hipLaunchKernelGGL(pack_u_kernel, dim3((NT_PROJ * KC_PROJ * 32 + 255) / 256),
                     dim3(256), 0, stream, U, Up);
  hipLaunchKernelGGL(lstm_kernel, dim3(B_ROWS / 16), dim3(320), 0, stream,
                     ids, emb, bl, Wp, hB);
  hipLaunchKernelGGL(proj_kernel, dim3(B_ROWS / 32), dim3(256), 0, stream,
                     hB, Up, b2, out);
}
